// GraphAttention_83700322664683
// MI455X (gfx1250) — compile-verified, matrix operand order checked by reading, not run on hardware
//
#include <hip/hip_runtime.h>

// ---------------------------------------------------------------------------
// GraphAttention fused for MI455X (gfx1250, wave32, WMMA + TDM).
//
// One workgroup (256 thr = 8 waves) per (b, n) node row:
//   Phase A: TDM tensor_load_to_lds DMAs e[b,n,:,:] (64KB fp32) into LDS with
//            hardware row padding (stride 132 floats -> conflict-free frags);
//            q row, FiLM rows, mask staged in parallel.
//   Phase B: WMMA GEMM  e @ We_mul / We_add  (K=128, bf16, f32 acc), fuse
//            sim = (1+e_mul*em)*(q*k*scale) + e_add*em  -> LDS bf16 [128x512]
//   Phase C: WMMA GEMM  film_y(sim) @ We_out (K=512), +bias, *em -> e_out
//   Phase D: channel-wise online softmax over m + attn@v + node FiLM
//   Phase E: out-row @ Wo matvec, +bias, *mask -> out
// Final kernel: masked means + y update.
// sim (134 MB if materialized in fp32) never touches HBM.
// ---------------------------------------------------------------------------

#define DEV static __device__ __forceinline__

typedef __attribute__((ext_vector_type(16))) __bf16 v16bf;
typedef __attribute__((ext_vector_type(8)))  float  v8f;
typedef __attribute__((ext_vector_type(4)))  unsigned int v4u;
typedef __attribute__((ext_vector_type(8)))  int    v8i;
typedef __attribute__((ext_vector_type(4)))  int    v4i;

constexpr int cB = 4, cN = 128, cNODE = 256, cEDGE = 128, cGLOB = 128, cHID = 512;
constexpr float cSCALE = 0.125f;     // D=64 -> D^-0.5
constexpr int SIMP  = 528;           // bf16 LDS stride for sim tile (bank pad)
constexpr int ESTRF = 132;           // f32 LDS stride for e tile (TDM pad: 128+4 dw)

// d_out layout: out[4*128*256] | e_out[4*128*128*128] | y_out[4*128]
constexpr size_t OUT_N  = (size_t)cB * cN * cNODE;          // 131072
constexpr size_t OUT_E  = (size_t)cB * cN * cN * cEDGE;     // 8388608

// ---- bf16 helpers (RNE) ----------------------------------------------------
DEV unsigned short f2bfu(float f) {
  unsigned u = __builtin_bit_cast(unsigned, f);
  u += 0x7FFFu + ((u >> 16) & 1u);
  return (unsigned short)(u >> 16);
}
DEV float bf2f(unsigned short h) {
  unsigned u = ((unsigned)h) << 16;
  return __builtin_bit_cast(float, u);
}
DEV unsigned pack2(float lo, float hi) {
  return (unsigned)f2bfu(lo) | ((unsigned)f2bfu(hi) << 16);
}
DEV v8f vzero() {
  v8f r;
#pragma unroll
  for (int i = 0; i < 8; ++i) r[i] = 0.f;
  return r;
}

// K offset of bf16 pair i (0..7) for 16-bit WMMA operands (ISA 7.12.2):
// VGPR0..3 hold K={0,1..6,7} (+8 for lanes 16..31), VGPR4..7 hold +16.
DEV int koff(int i, int lane) {
  int base = (lane & 16) ? 8 : 0;
  return (i < 4 ? 2 * i : 16 + 2 * (i - 4)) + base;
}

union Frag { v16bf v; unsigned u[8]; };

// Fragment gather from bf16 rows: row r0+(lane&15), contiguous pairs at
// scattered K. Used for B operands (pre-transposed weights, K-contiguous)
// only; Phase C builds its sim-tile A operand inline.
DEV v16bf load_frag_rows(const unsigned short* base, int stride, int r0, int kb, int lane) {
  int r = r0 + (lane & 15);
  const unsigned* row = (const unsigned*)(base + (size_t)r * stride);
  Frag f;
#pragma unroll
  for (int i = 0; i < 8; ++i) f.u[i] = row[(kb + koff(i, lane)) >> 1];
  return f.v;
}

// Fragment gather from padded f32 LDS e-tile (TDM-staged), cvt->bf16 on read.
DEV v16bf load_frag_f32(const float* base, int stride, int r0, int kb, int lane) {
  int r = r0 + (lane & 15);
  const float* row = base + (size_t)r * stride;
  Frag f;
#pragma unroll
  for (int i = 0; i < 8; ++i) {
    int k = kb + koff(i, lane);
    f.u[i] = pack2(row[k], row[k + 1]);
  }
  return f.v;
}

DEV v8f wmma_bf16(v16bf a, v16bf b, v8f c) {
  return __builtin_amdgcn_wmma_f32_16x16x32_bf16(false, a, false, b, (short)0, c,
                                                 false, false);
}

// ---- K0a: q/k/v projections (x @ W, masked) --------------------------------
__global__ __launch_bounds__(256) void ga_qkv(
    const float* __restrict__ x, const int* __restrict__ maskp,
    const float* __restrict__ Wq, const float* __restrict__ Wk,
    const float* __restrict__ Wv, float* __restrict__ q, float* __restrict__ k,
    float* __restrict__ v) {
  __shared__ float xr[cNODE];
  const int b = blockIdx.x >> 7, n = blockIdx.x & 127, tid = threadIdx.x;
  const float* xrow = x + ((size_t)(b * cN + n)) * cNODE;
  xr[tid] = xrow[tid];
  __syncthreads();
  const float xm = (float)maskp[b * cN + n];
  const size_t ob = ((size_t)(b * cN + n)) * cHID;
  for (int c = tid; c < cHID; c += 256) {
    float aq = 0.f, ak = 0.f, av = 0.f;
    for (int kk = 0; kk < cNODE; ++kk) {
      float xv = xr[kk];
      aq += xv * Wq[kk * cHID + c];
      ak += xv * Wk[kk * cHID + c];
      av += xv * Wv[kk * cHID + c];
    }
    q[ob + c] = aq * xm;
    k[ob + c] = ak * xm;
    v[ob + c] = av * xm;
  }
}

// ---- K0b: global FiLM vectors (1 + y@W_mul, y@W_add) -----------------------
__global__ __launch_bounds__(256) void ga_film(
    const float* __restrict__ y, const float* __restrict__ Wyx_mul,
    const float* __restrict__ Wyx_add, const float* __restrict__ Wye_mul,
    const float* __restrict__ Wye_add, float* __restrict__ yxm1,
    float* __restrict__ yxa1, float* __restrict__ yem1, float* __restrict__ yea1) {
  __shared__ float yr[cGLOB];
  const int b = blockIdx.x, tid = threadIdx.x;
  if (tid < cGLOB) yr[tid] = y[b * cGLOB + tid];
  __syncthreads();
  for (int c = tid; c < cHID; c += 256) {
    float a0 = 0.f, a1 = 0.f, a2 = 0.f, a3 = 0.f;
    for (int g = 0; g < cGLOB; ++g) {
      float yv = yr[g];
      a0 += yv * Wyx_mul[g * cHID + c];
      a1 += yv * Wyx_add[g * cHID + c];
      a2 += yv * Wye_mul[g * cHID + c];
      a3 += yv * Wye_add[g * cHID + c];
    }
    yxm1[b * cHID + c] = 1.f + a0;
    yxa1[b * cHID + c] = a1;
    yem1[b * cHID + c] = 1.f + a2;
    yea1[b * cHID + c] = a3;
  }
}

// ---- K0c: transpose+convert edge weights to bf16 (K-contiguous B operands) -
__global__ __launch_bounds__(256) void ga_wconv(
    const float* __restrict__ We_mul, const float* __restrict__ We_add,
    const float* __restrict__ We_out, unsigned short* __restrict__ WmulT,
    unsigned short* __restrict__ WaddT, unsigned short* __restrict__ WoutT) {
  const int idx = blockIdx.x * 256 + threadIdx.x;  // 3*65536 total
  const int which = idx >> 16, rem = idx & 65535;
  if (which == 0) {
    int c = rem >> 7, k = rem & 127;  // WmulT[c][k] = We_mul[k][c]
    WmulT[c * cEDGE + k] = f2bfu(We_mul[k * cHID + c]);
  } else if (which == 1) {
    int c = rem >> 7, k = rem & 127;
    WaddT[c * cEDGE + k] = f2bfu(We_add[k * cHID + c]);
  } else {
    int n = rem >> 9, k = rem & 511;  // WoutT[n][k] = We_out[k][n]
    WoutT[n * cHID + k] = f2bfu(We_out[k * cEDGE + n]);
  }
}

// ---- K1: fused per-(b,n) edge GEMMs + softmax + attention ------------------
__global__ __launch_bounds__(256) void ga_main(
    const float* __restrict__ e, const float* __restrict__ qws,
    const float* __restrict__ kws, const float* __restrict__ vws,
    const float* __restrict__ yxm1, const float* __restrict__ yxa1,
    const float* __restrict__ yem1, const float* __restrict__ yea1,
    const int* __restrict__ maskp, const unsigned short* __restrict__ WmulT,
    const unsigned short* __restrict__ WaddT,
    const unsigned short* __restrict__ WoutT, const float* __restrict__ be_out,
    const float* __restrict__ Wo, const float* __restrict__ bo,
    float* __restrict__ outp, float* __restrict__ eoutp) {
  extern __shared__ char smem[];
  unsigned short* simL = (unsigned short*)smem;   // [128][SIMP] bf16 sim tile
  float* eLf = (float*)(simL + 128 * SIMP);       // [128][ESTRF] f32 e tile (TDM)
  float* qrow = eLf + 128 * ESTRF;                // q[b,n,:] * scale
  float* yem = qrow + cHID;
  float* yea = yem + cHID;
  float* yxm = yea + cHID;
  float* yxa = yxm + cHID;
  float* orow = yxa + cHID;                       // attention output row
  int* mrow = (int*)(orow + cHID);                // mask[b,:]

  const int tid = threadIdx.x;
  const int b = blockIdx.x >> 7, n = blockIdx.x & 127;
  const int lane = tid & 31, wave = tid >> 5;
  const int m0 = wave * 16, half = lane >> 4, cl = lane & 15;
  const float xmn = (float)maskp[b * cN + n];

  // ---- Phase A: stage inputs -----------------------------------------------
  const float* e_bn = e + ((size_t)(b * cN + n)) * cN * cEDGE;
  __builtin_prefetch(WmulT, 0, 3);      // global_prefetch_b8 (L2-hot weights)
  __builtin_prefetch(WoutT, 0, 3);

#if __has_builtin(__builtin_amdgcn_tensor_load_to_lds)
  // Tensor Data Mover: DMA the 64KB e[b,n] tile into LDS. 1D tile of 16384
  // dwords; pad_enable inserts 4 dwords after every 128 -> LDS row stride 132.
  if (wave == 0) {
    const unsigned ldsaddr = (unsigned)(size_t)(void*)eLf;   // low 32b = LDS offset
    const unsigned long long ga = (unsigned long long)(size_t)(const void*)e_bn;
    v4u g0;
    g0[0] = 1u;                                       // count=1 valid descriptor
    g0[1] = ldsaddr;                                  // lds_addr
    g0[2] = (unsigned)(ga & 0xFFFFFFFFu);             // global_addr[31:0]
    g0[3] = (unsigned)((ga >> 32) & 0x01FFFFFFu)      // global_addr[56:32]
            | (2u << 30);                             // type=2 ("image")
    v8i g1;
    g1[0] = (int)((2u << 16)        // data_size = 4B
                  | (1u << 20)      // pad_enable
                  | (6u << 22)      // pad_interval: every 128 dwords
                  | (3u << 25));    // pad_amount: 4 dwords
    g1[1] = (int)(16384u << 16);    // tensor_dim0[15:0] = 16384
    g1[2] = (int)(1u << 16);        // tensor_dim0 hi=0 | tensor_dim1 = 1
    g1[3] = (int)(16384u << 16);    // tensor_dim1 hi=0 | tile_dim0 = 16384
    g1[4] = 1;                      // tile_dim1 = 1, tile_dim2 = 0
    g1[5] = 16384;                  // tensor_dim0_stride[31:0]
    g1[6] = 0;                      // stride hi | tensor_dim1_stride lo
    g1[7] = 0;
    v4i g2 = {0, 0, 0, 0}, g3 = {0, 0, 0, 0};
#if defined(__clang_major__) && (__clang_major__ >= 23)
    v8i g4 = {0, 0, 0, 0, 0, 0, 0, 0};
    __builtin_amdgcn_tensor_load_to_lds(g0, g1, g2, g3, g4, 0);
#else
    __builtin_amdgcn_tensor_load_to_lds(g0, g1, g2, g3, 0);
#endif
    __builtin_amdgcn_s_wait_tensorcnt(0);             // s_wait_tensorcnt 0
  }
#else
  for (int i = tid; i < cN * cEDGE; i += 256)
    eLf[(i >> 7) * ESTRF + (i & 127)] = e_bn[i];
#endif

  {
    const size_t fb = (size_t)b * cHID;
    const size_t qb = ((size_t)(b * cN + n)) * cHID;
    for (int i = tid; i < cHID; i += 256) {
      qrow[i] = qws[qb + i] * cSCALE;
      yem[i] = yem1[fb + i];
      yea[i] = yea1[fb + i];
      yxm[i] = yxm1[fb + i];
      yxa[i] = yxa1[fb + i];
    }
  }
  if (tid < cN) mrow[tid] = maskp[b * cN + tid];
  __syncthreads();

  // ---- Phase B: sim = (1 + em*e@We_mul) * q*k*scale + em*e@We_add ----------
  // Wave w owns m-tile w; A fragments (f32 LDS -> bf16) reused across 32 c-tiles.
  v16bf afr[4];
#pragma unroll
  for (int t = 0; t < 4; ++t) afr[t] = load_frag_f32(eLf, ESTRF, m0, 32 * t, lane);

  for (int ct = 0; ct < 32; ++ct) {
    v8f cm = vzero(), ca = vzero();
#pragma unroll
    for (int t = 0; t < 4; ++t) {
      v16bf bm = load_frag_rows(WmulT, cEDGE, ct * 16, 32 * t, lane);
      cm = wmma_bf16(afr[t], bm, cm);
      v16bf ba = load_frag_rows(WaddT, cEDGE, ct * 16, 32 * t, lane);
      ca = wmma_bf16(afr[t], ba, ca);
    }
    const int c = ct * 16 + cl;
    const float qs = qrow[c];
    const float* kcol = kws + ((size_t)b * cN) * cHID + c;
#pragma unroll
    for (int j = 0; j < 8; ++j) {
      int m = m0 + j + 8 * half;                    // C layout: VGPR j -> M
      float em = xmn * (float)mrow[m];
      float base = qs * kcol[(size_t)m * cHID];     // q,k already masked
      float s = (1.f + cm[j] * em) * base + ca[j] * em;
      simL[m * SIMP + c] = f2bfu(s);
    }
  }
  __syncthreads();

  // ---- Phase C: e_out = (film_y(sim) @ We_out + be_out) * em ---------------
  v8f acc[8];
#pragma unroll
  for (int i = 0; i < 8; ++i) acc[i] = vzero();

  for (int kb = 0; kb < 16; ++kb) {
    Frag af;  // film'd A fragment built from LDS sim (K dim = HID channel)
    {
      const int m = m0 + cl;
      const unsigned short* srow = simL + (size_t)m * SIMP;
#pragma unroll
      for (int i = 0; i < 8; ++i) {
        int k0 = kb * 32 + koff(i, lane);
        float lo = yem[k0] * bf2f(srow[k0]) + yea[k0];
        float hi = yem[k0 + 1] * bf2f(srow[k0 + 1]) + yea[k0 + 1];
        af.u[i] = pack2(lo, hi);
      }
    }
#pragma unroll
    for (int ct = 0; ct < 8; ++ct) {
      v16bf bf = load_frag_rows(WoutT, cHID, ct * 16, kb * 32, lane);
      acc[ct] = wmma_bf16(af.v, bf, acc[ct]);
    }
  }
  {
    float* eo = eoutp + ((size_t)(b * cN + n)) * cN * cEDGE;
#pragma unroll
    for (int ct = 0; ct < 8; ++ct) {
      const int c = ct * 16 + cl;
      const float bias = be_out[c];
#pragma unroll
      for (int j = 0; j < 8; ++j) {
        int m = m0 + j + 8 * half;
        float em = xmn * (float)mrow[m];
        eo[(size_t)m * cEDGE + c] = (acc[ct][j] + bias) * em;
      }
    }
  }
  __syncthreads();

  // ---- Phase D: channel-wise online softmax over m + attn @ v + node FiLM --
#pragma unroll
  for (int cc = 0; cc < 2; ++cc) {
    const int c = tid * 2 + cc;
    const float* vcol = vws + ((size_t)b * cN) * cHID + c;
    float mx = -__builtin_inff(), S = 0.f, A = 0.f;
    for (int m = 0; m < cN; ++m) {
      float s = mrow[m] ? bf2f(simL[m * SIMP + c]) : -__builtin_inff();
      if (s > mx) {
        float r = __expf(mx - s);   // exp(-inf)=0 on first hit: resets cleanly
        S *= r;
        A *= r;
        mx = s;
      }
      float p = (s == -__builtin_inff()) ? 0.f : __expf(s - mx);
      S += p;
      A += p * vcol[(size_t)m * cHID];
    }
    float o = A / S;
    orow[c] = yxm[c] * o + yxa[c];
  }
  __syncthreads();

  // ---- Phase E: out = (orow @ Wo + bo) * mask ------------------------------
  {
    float a = 0.f;
    for (int k0 = 0; k0 < cHID; ++k0) a += orow[k0] * Wo[k0 * cNODE + tid];
    outp[((size_t)(b * cN + n)) * cNODE + tid] = (a + bo[tid]) * xmn;
  }
}

// ---- K2: masked means + global y update ------------------------------------
__global__ __launch_bounds__(256) void ga_yupd(
    const float* __restrict__ outp, const float* __restrict__ eoutp,
    const float* __restrict__ y, const int* __restrict__ maskp,
    const float* __restrict__ Wyx_out, const float* __restrict__ byx_out,
    const float* __restrict__ Wye_out, const float* __restrict__ bye_out,
    float* __restrict__ youtp) {
  __shared__ float yx[cNODE];
  __shared__ float yeh[cEDGE];
  const int b = blockIdx.x, tid = threadIdx.x;
  float sn = 0.f;
  for (int nn = 0; nn < cN; ++nn) sn += (float)maskp[b * cN + nn];
  {
    float s = 0.f;
    for (int nn = 0; nn < cN; ++nn)
      s += outp[((size_t)(b * cN + nn)) * cNODE + tid];
    yx[tid] = s / sn;
  }
  if (tid < cEDGE) {
    float s = 0.f;
    const float* ep = eoutp + (size_t)b * cN * cN * cEDGE + tid;
    for (int r = 0; r < cN * cN; ++r) s += ep[(size_t)r * cEDGE];
    yeh[tid] = s / (sn * sn);
  }
  __syncthreads();
  if (tid < cGLOB) {
    float a = y[b * cGLOB + tid] + byx_out[tid] + bye_out[tid];
    for (int j = 0; j < cNODE; ++j) a += yx[j] * Wyx_out[j * cGLOB + tid];
    for (int j = 0; j < cEDGE; ++j) a += yeh[j] * Wye_out[j * cGLOB + tid];
    youtp[b * cGLOB + tid] = a;
  }
}

// ---------------------------------------------------------------------------
extern "C" void kernel_launch(void* const* d_in, const int* in_sizes, int n_in,
                              void* d_out, int out_size, void* d_ws,
                              size_t ws_size, hipStream_t stream) {
  const float* x = (const float*)d_in[0];
  const float* e = (const float*)d_in[1];
  const float* y = (const float*)d_in[2];
  const int* mask = (const int*)d_in[3];
  const float* Wq = (const float*)d_in[4];
  const float* Wk = (const float*)d_in[5];
  const float* Wv = (const float*)d_in[6];
  const float* Wo = (const float*)d_in[7];
  const float* bo = (const float*)d_in[8];
  const float* We_mul = (const float*)d_in[9];
  const float* We_add = (const float*)d_in[10];
  const float* We_out = (const float*)d_in[11];
  const float* be_out = (const float*)d_in[12];
  const float* Wyx_mul = (const float*)d_in[13];
  const float* Wyx_add = (const float*)d_in[14];
  const float* Wye_mul = (const float*)d_in[15];
  const float* Wye_add = (const float*)d_in[16];
  const float* Wyx_out = (const float*)d_in[17];
  const float* byx_out = (const float*)d_in[18];
  const float* Wye_out = (const float*)d_in[19];
  const float* bye_out = (const float*)d_in[20];

  float* out = (float*)d_out;
  float* eout = out + OUT_N;
  float* yout = out + OUT_N + OUT_E;

  // workspace carve (fp32): q|k|v (1 MB each), FiLM vectors, bf16 weights
  float* q = (float*)d_ws;
  float* kk = q + (size_t)cB * cN * cHID;       // 262144
  float* v = kk + (size_t)cB * cN * cHID;
  float* yxm1 = v + (size_t)cB * cN * cHID;
  float* yxa1 = yxm1 + cB * cHID;
  float* yem1 = yxa1 + cB * cHID;
  float* yea1 = yem1 + cB * cHID;
  unsigned short* WmulT = (unsigned short*)(yea1 + cB * cHID);  // [512][128]
  unsigned short* WaddT = WmulT + cHID * cEDGE;                 // [512][128]
  unsigned short* WoutT = WaddT + cHID * cEDGE;                 // [128][512]

  ga_qkv<<<dim3(cB * cN), dim3(256), 0, stream>>>(x, mask, Wq, Wk, Wv, q, kk, v);
  ga_film<<<dim3(cB), dim3(256), 0, stream>>>(y, Wyx_mul, Wyx_add, Wye_mul,
                                              Wye_add, yxm1, yxa1, yem1, yea1);
  ga_wconv<<<dim3(768), dim3(256), 0, stream>>>(We_mul, We_add, We_out, WmulT,
                                                WaddT, WoutT);

  const size_t shmem = (size_t)(128 * SIMP) * sizeof(unsigned short) +
                       (size_t)(128 * ESTRF) * sizeof(float) +
                       (size_t)(6 * cHID) * sizeof(float) + (size_t)cN * sizeof(int);
  ga_main<<<dim3(cB * cN), dim3(256), shmem, stream>>>(
      e, q, kk, v, yxm1, yxa1, yem1, yea1, mask, WmulT, WaddT, WoutT, be_out, Wo,
      bo, out, eout);

  ga_yupd<<<dim3(cB), dim3(256), 0, stream>>>(out, eout, y, mask, Wyx_out,
                                              byx_out, Wye_out, bye_out, yout);
}
